// PlaneNet_5695126634945
// MI455X (gfx1250) — matrix operand order repaired and checked by reading, not compiled, not run on hardware
//
#include <hip/hip_runtime.h>
#include <cmath>

// Problem constants (match reference)
constexpr int P  = 3;       // planes
constexpr int NN = 50000;   // nodes per plane  (= 3125 * 16, exact M tiles)
constexpr int E  = 400000;  // edges per plane
constexpr int C  = 5;       // classes
constexpr int F  = 80;      // node feature width
constexpr int F2 = 160;     // 2F concat width  (= 5 * 32 WMMA K-steps)
constexpr int PF = 64;      // planar feature / output width (= 4 * 16 N tiles, 2 * 32 K-steps)

typedef __attribute__((ext_vector_type(16))) _Float16 v16h;
typedef __attribute__((ext_vector_type(8)))  float    v8f;

// ---------------------------------------------------------------------------
// Kernel 0: zero the aggregation scratch (240 MB) each call (ws is poisoned)
// ---------------------------------------------------------------------------
__global__ __launch_bounds__(256) void zero_ws_kernel(float4* __restrict__ ws, long n4) {
    long i = (long)blockIdx.x * blockDim.x + threadIdx.x;
    long stride = (long)gridDim.x * blockDim.x;
    float4 z; z.x = 0.f; z.y = 0.f; z.z = 0.f; z.w = 0.f;
    for (; i < n4; i += stride) ws[i] = z;
}

// ---------------------------------------------------------------------------
// Kernel 1: edge messages. One wave32 per edge.
//   logits[c] = dot(cat(x_dst, x_src)[c], eW[c]) + eb[c]   (lane-strided + shfl reduce)
//   w = softmax_c(logits); aggr[dst,c,:] += w[c] * x_src[c,:]  (f32 atomics)
// ---------------------------------------------------------------------------
__global__ __launch_bounds__(256) void edge_kernel(
    const float* __restrict__ x, const int* __restrict__ ei,
    const float* __restrict__ eW, const float* __restrict__ eb,
    float* __restrict__ aggr)
{
    const int lane = threadIdx.x & 31;
    const int e    = blockIdx.x * 8 + (threadIdx.x >> 5);
    const int p    = blockIdx.y;
    if (e >= E) return;  // uniform per wave

    const int src = ei[((size_t)p * 2 + 0) * E + e];
    const int dst = ei[((size_t)p * 2 + 1) * E + e];

    const float* xi = x + ((size_t)p * NN + dst) * (C * F);
    const float* xj = x + ((size_t)p * NN + src) * (C * F);
    const float* wp = eW + (size_t)p * C * F2;

    float logit[C];
#pragma unroll
    for (int c = 0; c < C; ++c) {
        float part = 0.f;
        for (int f = lane; f < F2; f += 32) {
            float v = (f < F) ? xi[c * F + f] : xj[c * F + (f - F)];
            part += v * wp[c * F2 + f];
        }
#pragma unroll
        for (int off = 16; off >= 1; off >>= 1)
            part += __shfl_xor(part, off, 32);
        logit[c] = part + eb[p * C + c];
    }

    // softmax over class dim (all lanes hold identical values)
    float mx = logit[0];
#pragma unroll
    for (int c = 1; c < C; ++c) mx = fmaxf(mx, logit[c]);
    float wgt[C]; float s = 0.f;
#pragma unroll
    for (int c = 0; c < C; ++c) { wgt[c] = __expf(logit[c] - mx); s += wgt[c]; }
    float inv = 1.f / s;

    float* ag = aggr + ((size_t)p * NN + dst) * (C * F);
#pragma unroll
    for (int c = 0; c < C; ++c) {
        float wc = wgt[c] * inv;
        for (int f = lane; f < F; f += 32) {
#ifdef __HIP_PLATFORM_AMD__
            unsafeAtomicAdd(&ag[c * F + f], wc * xj[c * F + f]);
#else
            atomicAdd(&ag[c * F + f], wc * xj[c * F + f]);
#endif
        }
    }
}

// ---------------------------------------------------------------------------
// Kernel 2: node network via WMMA f32_16x16x32_f16.
// Grid (3125, C, P). Block = 128 threads = 4 waves; wave w owns output
// columns [16w, 16w+16). 16 rows per block (exact: 50000 = 3125*16).
// ---------------------------------------------------------------------------
__global__ __launch_bounds__(128) void node_kernel(
    const float* __restrict__ x, const float* __restrict__ aggr,
    const float* __restrict__ W1, const float* __restrict__ b1,
    const float* __restrict__ W2, const float* __restrict__ b2,
    float* __restrict__ out)
{
    __shared__ _Float16 sA[16][162];  // 16 x 160 cat tile, padded (81-dword row stride)
    __shared__ _Float16 sH[16][66];   // 16 x 64 hidden tile, padded (33-dword stride)

    const int tile = blockIdx.x;
    const int c    = blockIdx.y;
    const int p    = blockIdx.z;
    const int tid  = threadIdx.x;
    const int row0 = tile * 16;

    // ---- stage cat(x, aggr) rows into LDS as f16 (coalesced f32 loads) ----
    {
        const float* xb = x    + (size_t)p * NN * (C * F);
        const float* ab = aggr + (size_t)p * NN * (C * F);
        for (int idx = tid; idx < 16 * F2; idx += 128) {
            int r = idx / F2, f = idx - r * F2;
            size_t base = (size_t)(row0 + r) * (C * F) + (size_t)c * F;
            float v = (f < F) ? xb[base + f] : ab[base + (f - F)];
            sA[r][f] = (_Float16)v;
        }
    }
    __syncthreads();

    const int wv   = tid >> 5;
    const int lane = tid & 31;
    const int nloc = lane & 15;   // A: row M,  B: col N,  C/D: col N
    const int h    = lane >> 4;   // half-wave select
    const int col  = wv * 16 + nloc;

    // ---------------- layer 1: [16x160] @ [160x64] ----------------
    const float* W1c = W1 + (size_t)(p * C + c) * F2 * PF;
    v8f acc;
    {
        float bias = b1[(size_t)(p * C + c) * PF + col];
#pragma unroll
        for (int r = 0; r < 8; ++r) acc[r] = bias;
    }
#pragma unroll
    for (int kk = 0; kk < F2 / 32; ++kk) {
        const int k0 = kk * 32;
        v16h a, b;
#pragma unroll
        for (int t = 0; t < 8; ++t) {            // A: lane holds M=nloc, K split by h
            a[t]     = sA[nloc][k0 + 8 * h + t];
            a[8 + t] = sA[nloc][k0 + 16 + 8 * h + t];
        }
#pragma unroll
        for (int t = 0; t < 8; ++t) {            // B: lane holds N=col, K group by h
            b[t]     = (_Float16)W1c[(size_t)(k0 + 8 * h + t) * PF + col];
            b[8 + t] = (_Float16)W1c[(size_t)(k0 + 16 + 8 * h + t) * PF + col];
        }
        acc = __builtin_amdgcn_wmma_f32_16x16x32_f16(
            false, a, false, b, (short)0, acc, false, false);
    }
    // tanh -> hidden tile in LDS (f16)
#pragma unroll
    for (int r = 0; r < 8; ++r) {
        int m = r + 8 * h;                       // C/D: vgpr r + half-select
        sH[m][col] = (_Float16)tanhf(acc[r]);
    }
    __syncthreads();

    // ---------------- layer 2: [16x64] @ [64x64] ----------------
    const float* W2c = W2 + (size_t)(p * C + c) * PF * PF;
    v8f acc2;
    {
        float bias = b2[(size_t)(p * C + c) * PF + col];
#pragma unroll
        for (int r = 0; r < 8; ++r) acc2[r] = bias;
    }
#pragma unroll
    for (int kk = 0; kk < PF / 32; ++kk) {
        const int k0 = kk * 32;
        v16h a, b;
#pragma unroll
        for (int t = 0; t < 8; ++t) {
            a[t]     = sH[nloc][k0 + 8 * h + t];
            a[8 + t] = sH[nloc][k0 + 16 + 8 * h + t];
        }
#pragma unroll
        for (int t = 0; t < 8; ++t) {
            b[t]     = (_Float16)W2c[(size_t)(k0 + 8 * h + t) * PF + col];
            b[8 + t] = (_Float16)W2c[(size_t)(k0 + 16 + 8 * h + t) * PF + col];
        }
        acc2 = __builtin_amdgcn_wmma_f32_16x16x32_f16(
            false, a, false, b, (short)0, acc2, false, false);
    }

    float* ob = out + ((size_t)p * NN + row0) * (C * PF) + (size_t)c * PF;
#pragma unroll
    for (int r = 0; r < 8; ++r) {
        int m = r + 8 * h;
        ob[(size_t)m * (C * PF) + col] = tanhf(acc2[r]);
    }
}

// ---------------------------------------------------------------------------
extern "C" void kernel_launch(void* const* d_in, const int* in_sizes, int n_in,
                              void* d_out, int out_size, void* d_ws, size_t ws_size,
                              hipStream_t stream) {
    (void)in_sizes; (void)n_in; (void)out_size; (void)ws_size;
    const float* x  = (const float*)d_in[0];
    const int*   ei = (const int*)  d_in[1];
    const float* eW = (const float*)d_in[2];
    const float* eb = (const float*)d_in[3];
    const float* W1 = (const float*)d_in[4];
    const float* b1 = (const float*)d_in[5];
    const float* W2 = (const float*)d_in[6];
    const float* b2 = (const float*)d_in[7];
    float* out  = (float*)d_out;
    float* aggr = (float*)d_ws;   // needs P*NN*C*F floats = 240 MB

    const long nAggr = (long)P * NN * C * F;           // 60,000,000 (divisible by 4)
    zero_ws_kernel<<<4096, 256, 0, stream>>>((float4*)d_ws, nAggr / 4);

    edge_kernel<<<dim3((E + 7) / 8, P), 256, 0, stream>>>(x, ei, eW, eb, aggr);

    node_kernel<<<dim3(NN / 16, C, P), 128, 0, stream>>>(x, aggr, W1, b1, W2, b2, out);
}
